// Summerize_90555090469163
// MI455X (gfx1250) — matrix operand order repaired and checked by reading, not compiled, not run on hardware
//
#include <hip/hip_runtime.h>
#include <stdint.h>

typedef __attribute__((ext_vector_type(16))) _Float16 v16h;
typedef __attribute__((ext_vector_type(8)))  float    v8f;

#define BATCH 16
#define NROWS 50000
#define CH    128
#define TOPK  1024
#define TILES_PER_BATCH (NROWS / 16)          // 3125
#define NTILES (BATCH * TILES_PER_BATCH)      // 50000

// ---------------------------------------------------------------------------
// Kernel 1: scores via WMMA. One wave computes 16 rows x 128 ch per tile.
// D = A(16x32 f16 rows of inputs) x B(32x16 f16, p replicated over columns)
// accumulated f32 over 4 chunks of 32 channels. Every column of D equals the
// score vector, so lanes with N==0 (lanes 0 and 16) extract rows 0-7 / 8-15.
// ---------------------------------------------------------------------------
__global__ void score_wmma_kernel(const float* __restrict__ inp,
                                  const float* __restrict__ p,
                                  float* __restrict__ y)
{
    const int lane       = threadIdx.x & 31;
    const int waveInBlk  = threadIdx.x >> 5;
    const int wavesPerBlk = blockDim.x >> 5;
    const int gwave  = blockIdx.x * wavesPerBlk + waveInBlk;
    const int nwaves = gridDim.x * wavesPerBlk;

    const int m    = lane & 15;   // row within tile (A) / column (B)
    const int half = lane >> 4;

    // B fragments: B[k][n] = p[k] for all n. Dense f16 B 32x16 layout
    // (contiguous k order used here, same as A): lanes 0-15 hold K=0..15, lanes 16-31
    // hold K=16..31, packed 2 per VGPR.
    v16h bf[4];
#pragma unroll
    for (int c = 0; c < 4; ++c) {
#pragma unroll
        for (int v = 0; v < 8; ++v) {
            const int k = c * 32 + half * 16 + 2 * v;
            bf[c][2 * v]     = (_Float16)p[k];
            bf[c][2 * v + 1] = (_Float16)p[k + 1];
        }
    }

    for (int tile = gwave; tile < NTILES; tile += nwaves) {
        const int b  = tile / TILES_PER_BATCH;
        const int n0 = (tile - b * TILES_PER_BATCH) * 16;
        const float* row = inp + ((size_t)b * NROWS + n0 + m) * CH;

        v8f acc = {};
#pragma unroll
        for (int c = 0; c < 4; ++c) {
            // 16-bit A 16x32 layout (matches B): K=16*half+2v
            v16h a;
#pragma unroll
            for (int v = 0; v < 8; ++v) {
                const int k0 = c * 32 + half * 16 + 2 * v;
                const float2 f2 = *(const float2*)(row + k0);
                a[2 * v]     = (_Float16)f2.x;
                a[2 * v + 1] = (_Float16)f2.y;
            }
            acc = __builtin_amdgcn_wmma_f32_16x16x32_f16(
                /*neg_a=*/false, a, /*neg_b=*/false, bf[c],
                /*c_mod=*/(short)0, acc, /*reuse_a=*/false, /*reuse_b=*/false);
        }

        // lane 0 holds D[0..7][0], lane 16 holds D[8..15][0]
        if ((lane & 15) == 0) {
            float* yo = y + (size_t)b * NROWS + n0 + half * 8;
#pragma unroll
            for (int r = 0; r < 8; ++r) yo[r] = acc[r];
        }
    }
}

// ---------------------------------------------------------------------------
// Kernel 2: per-batch exact top-K (radix select over order-preserving keys,
// LDS compaction, 1024-wide bitonic sort descending, stable on index).
// ---------------------------------------------------------------------------
__device__ __forceinline__ uint32_t fkey(float f)
{
    uint32_t u = __float_as_uint(f);
    return (u & 0x80000000u) ? ~u : (u | 0x80000000u);
}

__global__ void __launch_bounds__(1024)
topk_kernel(const float* __restrict__ y, uint32_t* __restrict__ selIdx)
{
    __shared__ uint32_t hist[256];
    __shared__ unsigned long long pairs[TOPK];
    __shared__ uint32_t s_prefix, s_need, s_cnt, s_tie;

    const int tid = threadIdx.x;          // blockDim.x == 1024
    const int b   = blockIdx.x;
    const float* yb = y + (size_t)b * NROWS;

    if (tid == 0) { s_prefix = 0u; s_need = TOPK; }
    __syncthreads();

    // 4-pass byte-wise radix select (MSB first) for the K-th largest key
    for (int pos = 3; pos >= 0; --pos) {
        if (tid < 256) hist[tid] = 0u;
        __syncthreads();
        const uint32_t prefix = s_prefix;
        const uint32_t hiMask = (pos == 3) ? 0u : (0xFFFFFFFFu << (8 * (pos + 1)));
        for (int i = tid; i < NROWS; i += 1024) {
            const uint32_t key = fkey(yb[i]);
            if (((key ^ prefix) & hiMask) == 0u)
                atomicAdd(&hist[(key >> (8 * pos)) & 0xFFu], 1u);
        }
        __syncthreads();
        if (tid == 0) {
            uint32_t cum = 0, need = s_need;
            int v = 255;
            for (; v > 0; --v) {
                const uint32_t c = hist[v];
                if (cum + c >= need) break;
                cum += c;
            }
            s_prefix = prefix | ((uint32_t)v << (8 * pos));
            s_need   = need - cum;
        }
        __syncthreads();
    }

    const uint32_t T     = s_prefix;   // exact threshold key
    const uint32_t nties = s_need;     // # of elements == T to keep
    if (tid == 0) { s_cnt = 0u; s_tie = 0u; }
    pairs[tid] = 0ull;
    __syncthreads();

    // Compact the K winners into LDS
    for (int i = tid; i < NROWS; i += 1024) {
        const uint32_t key = fkey(yb[i]);
        bool take = false;
        if (key > T)       take = true;
        else if (key == T) take = (atomicAdd(&s_tie, 1u) < nties);
        if (take) {
            const uint32_t slot = atomicAdd(&s_cnt, 1u);
            if (slot < TOPK)
                pairs[slot] = ((unsigned long long)key << 32) |
                              (uint32_t)(~(uint32_t)i);   // ~idx: low idx sorts first
        }
    }
    __syncthreads();

    // Bitonic sort, descending (=> score desc, index asc on ties)
    for (unsigned k = 2; k <= TOPK; k <<= 1) {
        for (unsigned j = k >> 1; j > 0; j >>= 1) {
            __syncthreads();
            const unsigned i = (unsigned)tid, ixj = i ^ j;
            if (ixj > i) {
                const unsigned long long a = pairs[i], c = pairs[ixj];
                const bool up = ((i & k) == 0);
                if (up ? (a < c) : (a > c)) { pairs[i] = c; pairs[ixj] = a; }
            }
        }
    }
    __syncthreads();

    selIdx[(size_t)b * TOPK + tid] = ~(uint32_t)(pairs[tid] & 0xFFFFFFFFull);
}

// ---------------------------------------------------------------------------
// Kernel 3: gather + exact f32 rescore + tanh scale. One wave per output row.
// ---------------------------------------------------------------------------
__global__ void gather_kernel(const float* __restrict__ inp,
                              const float* __restrict__ p,
                              const uint32_t* __restrict__ selIdx,
                              float* __restrict__ out)
{
    const int lane = threadIdx.x & 31;
    const int wave = (int)((blockIdx.x * blockDim.x + threadIdx.x) >> 5);
    const int b = wave >> 10;          // wave / 1024
    const int k = wave & 1023;

    const float4 p4 = *(const float4*)(p + lane * 4);
    float nrm2 = p4.x * p4.x + p4.y * p4.y + p4.z * p4.z + p4.w * p4.w;
#pragma unroll
    for (int off = 16; off > 0; off >>= 1) nrm2 += __shfl_xor(nrm2, off, 32);
    const float invn = rsqrtf(nrm2);

    uint32_t idx = selIdx[(size_t)b * TOPK + k];
    if (idx >= NROWS) idx = 0;   // safety clamp

    const float4 x4 = *(const float4*)(inp + ((size_t)b * NROWS + idx) * CH + lane * 4);
    float d = x4.x * p4.x + x4.y * p4.y + x4.z * p4.z + x4.w * p4.w;
#pragma unroll
    for (int off = 16; off > 0; off >>= 1) d += __shfl_xor(d, off, 32);
    const float s = tanhf(d * invn);

    float4 o;
    o.x = x4.x * s; o.y = x4.y * s; o.z = x4.z * s; o.w = x4.w * s;
    *(float4*)(out + ((size_t)b * TOPK + k) * CH + lane * 4) = o;
}

// ---------------------------------------------------------------------------
extern "C" void kernel_launch(void* const* d_in, const int* in_sizes, int n_in,
                              void* d_out, int out_size, void* d_ws, size_t ws_size,
                              hipStream_t stream)
{
    (void)in_sizes; (void)n_in; (void)out_size; (void)ws_size;

    const float* inp = (const float*)d_in[0];   // [16, 50000, 128] f32
    const float* p   = (const float*)d_in[1];   // [128] f32
    float* out = (float*)d_out;                 // [16, 1024, 128] f32

    float*    y      = (float*)d_ws;                                     // 3.2 MB
    uint32_t* selIdx = (uint32_t*)((char*)d_ws +
                        (size_t)BATCH * NROWS * sizeof(float));          // 64 KB

    // 1024 blocks x 8 waves = 8192 waves grid-striding 50000 tiles
    score_wmma_kernel<<<1024, 256, 0, stream>>>(inp, p, y);
    topk_kernel<<<BATCH, 1024, 0, stream>>>(y, selIdx);
    gather_kernel<<<(BATCH * TOPK) / 8, 256, 0, stream>>>(inp, p, selIdx, out);
}
